// MultiHeadAttention_16509854286463
// MI455X (gfx1250) — compile-verified, matrix operand order checked by reading, not run on hardware
//
#include <hip/hip_runtime.h>
#include <hip/hip_bf16.h>

typedef __attribute__((ext_vector_type(16))) _Float16 v16h;
typedef __attribute__((ext_vector_type(8)))  _Float16 v8h;
typedef __attribute__((ext_vector_type(8)))  float    v8f;
typedef __attribute__((ext_vector_type(4)))  unsigned int v4u;
typedef __attribute__((ext_vector_type(8)))  int      v8i;
typedef __attribute__((ext_vector_type(4)))  int      v4i;

#define B_  4
#define N_  2048
#define D_  1024
#define H_  16
#define HD_ 64
#define RD_ 32

#if defined(__has_builtin)
#if __has_builtin(__builtin_amdgcn_tensor_load_to_lds)
#define HAVE_TDM 1
#endif
#endif
#ifndef HAVE_TDM
#define HAVE_TDM 0
#endif

#if __has_include(<hip/amd_detail/amd_gfx1250_TDM.h>)
#define TDM_6ARG 1   // therock/clang-23 headers -> 6-arg builtin
#else
#define TDM_6ARG 0   // ROCm 7.2 clang-22 -> 5-arg builtin
#endif

// ---------------------------------------------------------------------------
// Fragment loader matching the CDNA5 16-bit A/B WMMA VGPR layout:
//   lanes 0-15  : row = base+lane,     halves K = kb..kb+7  and kb+16..kb+23
//   lanes 16-31 : row = base+lane-16,  halves K = kb+8..kb+15 and kb+24..kb+31
// Caller passes kb = k0 + (lane>>4)*8 so both groups are handled uniformly.
// ---------------------------------------------------------------------------
__device__ __forceinline__ v16h load_frag(const _Float16* __restrict__ p,
                                          int row, int kb, int ld) {
  const _Float16* base = p + (size_t)row * ld + kb;
  v8h lo = *(const v8h*)(base);
  v8h hi = *(const v8h*)(base + 16);
  v16h r;
#pragma unroll
  for (int i = 0; i < 8; ++i) { r[i] = lo[i]; r[i + 8] = hi[i]; }
  return r;
}

__device__ __forceinline__ v8f wmma_f16(v16h a, v16h b, v8f c) {
  return __builtin_amdgcn_wmma_f32_16x16x32_f16(false, a, false, b,
                                                (short)0, c, false, false);
}

#if HAVE_TDM
// ---------------------------------------------------------------------------
// Issue a TDM 2-D tile load: global (row-major, row_stride elems) -> LDS packed
// [tile_rows][tile_cols] f16.  D# per ISA 8.3/8.4: count=1, type=2, data_size=2B.
// One call per wave (EXEC ignored); tracked by TENSORcnt.
// ---------------------------------------------------------------------------
__device__ __forceinline__ void tdm_load_2d(unsigned lds_off,
                                            const _Float16* gptr,
                                            unsigned tile_rows, unsigned tile_cols,
                                            unsigned tensor_d0, unsigned tensor_d1,
                                            unsigned row_stride) {
  unsigned long long ga = (unsigned long long)(size_t)gptr;
  v4u g0;
  g0[0] = 1u;                                              // count=1, user mode
  g0[1] = lds_off;                                         // LDS byte address
  g0[2] = (unsigned)(ga & 0xFFFFFFFFu);                    // global_addr[31:0]
  g0[3] = (unsigned)((ga >> 32) & 0x1FFFFFFu) | (2u << 30);// addr[56:32] | type=2
  v8i g1;
  g1[0] = (int)(1u << 16);                                 // data_size=1 (2 bytes)
  g1[1] = (int)((tensor_d0 & 0xFFFFu) << 16);              // tensor_dim0[15:0]
  g1[2] = (int)((tensor_d0 >> 16) | ((tensor_d1 & 0xFFFFu) << 16));
  g1[3] = (int)((tensor_d1 >> 16) | (tile_cols << 16));    // tile_dim0
  g1[4] = (int)(tile_rows & 0xFFFFu);                      // tile_dim1 (tile_dim2=0)
  g1[5] = (int)row_stride;                                 // tensor_dim0_stride[31:0]
  g1[6] = 0;                                               // stride hi / dim1_stride lo
  g1[7] = 0;
  v4i z4 = {0, 0, 0, 0};
#if TDM_6ARG
  v8i z8 = {0, 0, 0, 0, 0, 0, 0, 0};
  __builtin_amdgcn_tensor_load_to_lds(g0, g1, z4, z4, z8, 0);
#else
  __builtin_amdgcn_tensor_load_to_lds(g0, g1, z4, z4, 0);
#endif
}
#endif

// ---------------------------------------------------------------------------
// Converters
// ---------------------------------------------------------------------------
__global__ void cvt_f16(const float* __restrict__ s, _Float16* __restrict__ d, int n) {
  int i = blockIdx.x * 256 + threadIdx.x;
  if (i < n) d[i] = (_Float16)s[i];
}

// d[n][k] = (f16) s[k][n]  (store W transposed for contiguous B-fragments)
__global__ void transpose_f16(const float* __restrict__ s, _Float16* __restrict__ d) {
  int i = blockIdx.x * 256 + threadIdx.x;
  int nrow = i >> 10;
  int k    = i & (D_ - 1);
  d[i] = (_Float16)s[k * D_ + nrow];
}

// ---------------------------------------------------------------------------
// WMMA GEMM:  C[8192 x 1024] = A[8192 x 1024] * Bt^T  (+ bias, fused epilogue)
// MODE 0: output proj -> f32 out
// MODE 1: Q: *HD^-0.5, RoPE, f16 out [B,H,N,64]
// MODE 2: K: RoPE, f16 out [B,H,N,64]
// MODE 3: V: f16 out transposed [B,H,64,N]
// Block tile 128(M) x 128(N), 256 thr = 8 waves (4 along M x 2 along N),
// wave tile 32x64 = 8 WMMA per K-step.  A/B tiles double-buffered in LDS via
// the Tensor Data Mover (wave 0 issues, TENSORcnt + barrier publishes).
// ---------------------------------------------------------------------------
template <int MODE>
__global__ __launch_bounds__(256) void gemm_proj(
    const _Float16* __restrict__ A, const _Float16* __restrict__ Bt,
    const float* __restrict__ bias, const float* __restrict__ pe,
    _Float16* __restrict__ outH, float* __restrict__ outF) {
  int tid  = threadIdx.x;
  int wave = tid >> 5;
  int ln   = tid & 31;
  int grp  = ln >> 4;
  int lg   = ln & 15;
  int wm   = wave >> 1;                 // 0..3 -> M offset wm*32
  int wn   = wave & 1;                  // 0..1 -> N offset wn*64
  int m0 = blockIdx.x * 128;
  int n0 = blockIdx.y * 128;

  const _Float16* Ab = A  + (size_t)m0 * D_;   // block A origin (rows m0..m0+127)
  const _Float16* Bb = Bt + (size_t)n0 * D_;   // block B origin (rows n0..n0+127)

  v8f acc[2][4] = {};

#if HAVE_TDM
  __shared__ __align__(16) _Float16 sA[2][128 * 32];
  __shared__ __align__(16) _Float16 sB[2][128 * 32];
  bool issuer = (tid < 32);
  unsigned ldsA[2], ldsB[2];
  ldsA[0] = (unsigned)(size_t)(void*)&sA[0][0];
  ldsA[1] = (unsigned)(size_t)(void*)&sA[1][0];
  ldsB[0] = (unsigned)(size_t)(void*)&sB[0][0];
  ldsB[1] = (unsigned)(size_t)(void*)&sB[1][0];

  if (issuer) {
    tdm_load_2d(ldsA[0], Ab + 0, 128, 32, D_, B_ * N_, D_);
    tdm_load_2d(ldsB[0], Bb + 0, 128, 32, D_, D_,      D_);
  }
  for (int k0 = 0; k0 < D_; k0 += 32) {
    int cur = (k0 >> 5) & 1;
    if (issuer) {
      if (k0 + 32 < D_) {
        tdm_load_2d(ldsA[cur ^ 1], Ab + (k0 + 32), 128, 32, D_, B_ * N_, D_);
        tdm_load_2d(ldsB[cur ^ 1], Bb + (k0 + 32), 128, 32, D_, D_,      D_);
        __builtin_amdgcn_s_wait_tensorcnt(2);   // pair for `cur` complete
      } else {
        __builtin_amdgcn_s_wait_tensorcnt(0);
      }
    }
    __syncthreads();                            // buf[cur] visible to all waves
    const _Float16* tA = &sA[cur][0];
    const _Float16* tB = &sB[cur][0];
    v16h a0 = load_frag(tA, wm * 32 +  0 + lg, grp * 8, 32);
    v16h a1 = load_frag(tA, wm * 32 + 16 + lg, grp * 8, 32);
#pragma unroll
    for (int t = 0; t < 4; ++t) {
      v16h b = load_frag(tB, wn * 64 + t * 16 + lg, grp * 8, 32);
      acc[0][t] = wmma_f16(a0, b, acc[0][t]);
      acc[1][t] = wmma_f16(a1, b, acc[1][t]);
    }
    __syncthreads();                            // done reading buf[cur]
  }
#else
  // Fallback: direct-global fragments (no TDM builtin on this toolchain)
  for (int k0 = 0; k0 < D_; k0 += 32) {
    v16h a0 = load_frag(Ab, wm * 32 +  0 + lg, k0 + grp * 8, D_);
    v16h a1 = load_frag(Ab, wm * 32 + 16 + lg, k0 + grp * 8, D_);
#pragma unroll
    for (int t = 0; t < 4; ++t) {
      v16h b = load_frag(Bb, wn * 64 + t * 16 + lg, k0 + grp * 8, D_);
      acc[0][t] = wmma_f16(a0, b, acc[0][t]);
      acc[1][t] = wmma_f16(a1, b, acc[1][t]);
    }
  }
#endif

  // ---- fused epilogue -----------------------------------------------------
#pragma unroll
  for (int s = 0; s < 2; ++s) {
#pragma unroll
    for (int t = 0; t < 4; ++t) {
      int col = n0 + wn * 64 + t * 16 + lg;   // C/D layout: col striped on lanes
      float bv = bias[col];
#pragma unroll
      for (int r = 0; r < 8; ++r) {
        int row = m0 + wm * 32 + s * 16 + grp * 8 + r;
        float val = acc[s][t][r] + bv;
        if (MODE == 1) val *= 0.125f;         // HD^-0.5
        if (MODE == 1 || MODE == 2) {
          int ntok = row & (N_ - 1);
          int hd   = col & (HD_ - 1);
          // rotate-half partner = adjacent column = adjacent lane (same row)
          float partner = __shfl_xor(val, 1, 32);
          if (hd < RD_) {
            float ang = pe[ntok * RD_ + hd];
            float cs = __cosf(ang), sn = __sinf(ang);
            val = (col & 1) ? (val * cs + partner * sn)
                            : (val * cs - partner * sn);
          }
        }
        if (MODE == 0) {
          outF[(size_t)row * D_ + col] = val;
        } else {
          int b = row >> 11, ntok = row & (N_ - 1);
          int h = col >> 6,  hd   = col & 63;
          if (MODE == 3)  // V transposed per head: [b,h][hd][n]
            outH[((size_t)(b * H_ + h) * HD_ + hd) * N_ + ntok] = (_Float16)val;
          else            // Q/K: [b,h][n][hd]
            outH[((size_t)(b * H_ + h) * N_ + ntok) * HD_ + hd] = (_Float16)val;
        }
      }
    }
  }
}

// ---------------------------------------------------------------------------
// Flash attention: grid (N/64, H, B), 128 thr = 4 waves, each wave owns 16 rows.
// Online softmax over 32-key blocks; P staged via LDS (C-layout -> A-layout).
// ---------------------------------------------------------------------------
__global__ __launch_bounds__(128) void attn_flash(
    const _Float16* __restrict__ Qh, const _Float16* __restrict__ Kh,
    const _Float16* __restrict__ Vt, const unsigned char* __restrict__ pad,
    _Float16* __restrict__ Oc) {
  __shared__ __align__(16) _Float16 lds[4 * 16 * 32];
  int tid  = threadIdx.x;
  int wave = tid >> 5;
  int ln   = tid & 31;
  int grp  = ln >> 4;
  int lg   = ln & 15;
  int i0 = blockIdx.x * 64;
  int h  = blockIdx.y;
  int b  = blockIdx.z;
  const _Float16* Q = Qh + (size_t)(b * H_ + h) * N_ * HD_;
  const _Float16* K = Kh + (size_t)(b * H_ + h) * N_ * HD_;
  const _Float16* V = Vt + (size_t)(b * H_ + h) * HD_ * N_;
  const unsigned char* pm = pad + (size_t)b * N_;
  _Float16* P = lds + wave * (16 * 32);

  int rbase = i0 + wave * 16;
  v16h qa0 = load_frag(Q, rbase + lg, 0  + grp * 8, HD_);
  v16h qa1 = load_frag(Q, rbase + lg, 32 + grp * 8, HD_);
  v8f o0 = {}, o1 = {}, o2 = {}, o3 = {};
  float m_[8], l_[8];
#pragma unroll
  for (int r = 0; r < 8; ++r) { m_[r] = -1e30f; l_[r] = 0.f; }

  int jend = rbase + 16;                       // causal: keys < rbase+16
  for (int j0 = 0; j0 < jend; j0 += 32) {
    // ---- S = Q K^T (two 16x16 tiles, K-dim 64) ----
    v8f s0 = {}, s1 = {};
    {
      v16h kb = load_frag(K, j0 + lg, grp * 8, HD_);
      s0 = wmma_f16(qa0, kb, s0);
      kb = load_frag(K, j0 + lg, 32 + grp * 8, HD_);
      s0 = wmma_f16(qa1, kb, s0);
      kb = load_frag(K, j0 + 16 + lg, grp * 8, HD_);
      s1 = wmma_f16(qa0, kb, s1);
      kb = load_frag(K, j0 + 16 + lg, 32 + grp * 8, HD_);
      s1 = wmma_f16(qa1, kb, s1);
    }
    int jj0 = j0 + lg, jj1 = j0 + 16 + lg;
    bool pad0 = pm[jj0] != 0;
    bool pad1 = pm[jj1] != 0;
    // ---- masked online softmax (row lives across 16 lanes at element r) ----
#pragma unroll
    for (int r = 0; r < 8; ++r) {
      int ii = rbase + grp * 8 + r;
      float v0 = (jj0 > ii || pad0) ? -1e30f : s0[r];
      float v1 = (jj1 > ii || pad1) ? -1e30f : s1[r];
      float mx = fmaxf(v0, v1);
#pragma unroll
      for (int off = 1; off < 16; off <<= 1) mx = fmaxf(mx, __shfl_xor(mx, off, 32));
      float nm = fmaxf(m_[r], mx);
      float al = __expf(m_[r] - nm);
      m_[r] = nm;
      float p0 = __expf(v0 - nm);
      float p1 = __expf(v1 - nm);
      float rs = p0 + p1;
#pragma unroll
      for (int off = 1; off < 16; off <<= 1) rs += __shfl_xor(rs, off, 32);
      l_[r] = l_[r] * al + rs;
      o0[r] *= al; o1[r] *= al; o2[r] *= al; o3[r] *= al;
      P[(grp * 8 + r) * 32 + lg]      = (_Float16)p0;
      P[(grp * 8 + r) * 32 + 16 + lg] = (_Float16)p1;
    }
    asm volatile("s_wait_dscnt 0" ::: "memory");   // our LDS stores -> our reads
    // ---- O += P V (A-fragment of P from LDS, B from transposed V) ----
    v16h pa = load_frag(P, lg, grp * 8, 32);
    v16h vb = load_frag(V, 0 * 16 + lg, j0 + grp * 8, N_);
    o0 = wmma_f16(pa, vb, o0);
    vb = load_frag(V, 1 * 16 + lg, j0 + grp * 8, N_);
    o1 = wmma_f16(pa, vb, o1);
    vb = load_frag(V, 2 * 16 + lg, j0 + grp * 8, N_);
    o2 = wmma_f16(pa, vb, o2);
    vb = load_frag(V, 3 * 16 + lg, j0 + grp * 8, N_);
    o3 = wmma_f16(pa, vb, o3);
  }

#pragma unroll
  for (int r = 0; r < 8; ++r) {
    int ii = rbase + grp * 8 + r;
    float inv = 1.0f / l_[r];
    size_t off = ((size_t)b * N_ + ii) * D_ + h * HD_;
    Oc[off + 0 * 16 + lg] = (_Float16)(o0[r] * inv);
    Oc[off + 1 * 16 + lg] = (_Float16)(o1[r] * inv);
    Oc[off + 2 * 16 + lg] = (_Float16)(o2[r] * inv);
    Oc[off + 3 * 16 + lg] = (_Float16)(o3[r] * inv);
  }
}

// ---------------------------------------------------------------------------
// Host launcher
// ---------------------------------------------------------------------------
extern "C" void kernel_launch(void* const* d_in, const int* in_sizes, int n_in,
                              void* d_out, int out_size, void* d_ws, size_t ws_size,
                              hipStream_t stream) {
  (void)in_sizes; (void)n_in; (void)out_size; (void)ws_size;
  const float* x_q  = (const float*)d_in[0];
  const float* x_kv = (const float*)d_in[1];
  const float* pe   = (const float*)d_in[2];
  const float* Wq   = (const float*)d_in[3];
  const float* bq   = (const float*)d_in[4];
  const float* Wk   = (const float*)d_in[5];
  const float* bk   = (const float*)d_in[6];
  const float* Wv   = (const float*)d_in[7];
  const float* bv   = (const float*)d_in[8];
  const float* Wo   = (const float*)d_in[9];
  const float* bo   = (const float*)d_in[10];
  const unsigned char* pad = (const unsigned char*)d_in[11];
  float* out = (float*)d_out;

  char* w = (char*)d_ws;
  const size_t MB = 1ull << 20;
  _Float16* Xq  = (_Float16*)(w + 0 * MB);    // 16 MB
  _Float16* Xkv = (_Float16*)(w + 16 * MB);   // 16 MB
  _Float16* Wqt = (_Float16*)(w + 32 * MB);   //  2 MB
  _Float16* Wkt = (_Float16*)(w + 34 * MB);   //  2 MB
  _Float16* Wvt = (_Float16*)(w + 36 * MB);   //  2 MB
  _Float16* Wot = (_Float16*)(w + 38 * MB);   //  2 MB
  _Float16* Qh  = (_Float16*)(w + 40 * MB);   // 16 MB
  _Float16* Kh  = (_Float16*)(w + 56 * MB);   // 16 MB
  _Float16* Vt  = (_Float16*)(w + 72 * MB);   // 16 MB
  _Float16* Oc  = (_Float16*)(w + 88 * MB);   // 16 MB (end: 104 MB)

  int nX = B_ * N_ * D_;
  cvt_f16<<<(nX + 255) / 256, 256, 0, stream>>>(x_q,  Xq,  nX);
  cvt_f16<<<(nX + 255) / 256, 256, 0, stream>>>(x_kv, Xkv, nX);
  int nW = D_ * D_;
  transpose_f16<<<nW / 256, 256, 0, stream>>>(Wq, Wqt);
  transpose_f16<<<nW / 256, 256, 0, stream>>>(Wk, Wkt);
  transpose_f16<<<nW / 256, 256, 0, stream>>>(Wv, Wvt);
  transpose_f16<<<nW / 256, 256, 0, stream>>>(Wo, Wot);

  dim3 gg((B_ * N_) / 128, D_ / 128);
  gemm_proj<1><<<gg, 256, 0, stream>>>(Xq,  Wqt, bq, pe, Qh, nullptr);
  gemm_proj<2><<<gg, 256, 0, stream>>>(Xkv, Wkt, bk, pe, Kh, nullptr);
  gemm_proj<3><<<gg, 256, 0, stream>>>(Xkv, Wvt, bv, nullptr, Vt, nullptr);

  attn_flash<<<dim3(N_ / 64, H_, B_), 128, 0, stream>>>(Qh, Kh, Vt, pad, Oc);

  gemm_proj<0><<<gg, 256, 0, stream>>>(Oc, Wot, bo, nullptr, nullptr, out);
}
